// MQAttention_8040178778720
// MI455X (gfx1250) — matrix operand order repaired and checked by reading, not compiled, not run on hardware
//
#include <hip/hip_runtime.h>
#include <hip/hip_bf16.h>

// ---------------------------------------------------------------------------
// Types for CDNA5 WMMA (wave32): v_wmma_f32_16x16x32_bf16
// ---------------------------------------------------------------------------
typedef __bf16 bf16;
typedef __attribute__((ext_vector_type(16))) __bf16 v16bf;
typedef __attribute__((ext_vector_type(8)))  __bf16 v8bf;
typedef __attribute__((ext_vector_type(8)))  float  v8f;

#define DEVI __device__ __forceinline__

union BF16Frag { v16bf v; v8bf h[2]; };

// A-matrix 16x32 bf16 fragment from row-major A (lda in elements).
// Layout: lanes 0-15 -> M=lane, elems 0..15 = K 0..15
//         lanes 16-31 -> M=lane-16, K 16..31.
DEVI v16bf load_a_frag(const bf16* A, int lda, int row0, int k0, int lane) {
  const int half = lane >> 4;
  const bf16* p = A + (size_t)(row0 + (lane & 15)) * lda + (k0 + half * 16);
  BF16Frag f;
  f.h[0] = *(const v8bf*)(p);
  f.h[1] = *(const v8bf*)(p + 8);
  return f.v;
}

// B-matrix 32x16 bf16 fragment, sourced from row-major B^T (N x K, ldb=K elems).
// Layout (contiguous k, same order as load_a_frag): lanes 0-15 hold K=0..15 (col = lane), lanes 16-31 hold K=16..31.
DEVI v16bf load_bt_frag(const bf16* BT, int ldb, int col0, int k0, int lane) {
  const bf16* p = BT + (size_t)(col0 + (lane & 15)) * ldb + (k0 + (lane >> 4) * 16);
  BF16Frag f;
  f.h[0] = *(const v8bf*)(p);
  f.h[1] = *(const v8bf*)(p + 8);
  return f.v;
}

DEVI v8f wmma_bf16(v16bf a, v16bf b, v8f c) {
  return __builtin_amdgcn_wmma_f32_16x16x32_bf16(false, a, false, b, (short)0, c,
                                                 false, false);
}

// ---------------------------------------------------------------------------
// Problem constants
// ---------------------------------------------------------------------------
static constexpr int FEAT = 1024;
static constexpr int TSEQ = 2048;
static constexpr int BATCH = 2;
static constexpr int HEADS = 16;
static constexpr int HDIM = 64;
static constexpr int ROWS = BATCH * TSEQ;   // 4096

// ---------------------------------------------------------------------------
// Kernel 1: LayerNorm -> bf16
// ---------------------------------------------------------------------------
__global__ void ln_kernel(const float* __restrict__ x, const float* __restrict__ g,
                          const float* __restrict__ be, bf16* __restrict__ xn) {
  const int row = blockIdx.x;
  const float* xr = x + (size_t)row * FEAT;
  float s = 0.f, s2 = 0.f;
  for (int i = threadIdx.x; i < FEAT; i += blockDim.x) {
    float v = xr[i];
    s += v;
    s2 += v * v;
  }
#pragma unroll
  for (int m = 16; m > 0; m >>= 1) {
    s += __shfl_xor(s, m, 32);
    s2 += __shfl_xor(s2, m, 32);
  }
  __shared__ float ss[8], ss2[8];
  const int wid = threadIdx.x >> 5;
  if ((threadIdx.x & 31) == 0) { ss[wid] = s; ss2[wid] = s2; }
  __syncthreads();
  s = 0.f; s2 = 0.f;
#pragma unroll
  for (int w = 0; w < 8; ++w) { s += ss[w]; s2 += ss2[w]; }
  const float mu = s * (1.f / FEAT);
  const float var = s2 * (1.f / FEAT) - mu * mu;
  const float rstd = rsqrtf(var + 1e-5f);
  bf16* xo = xn + (size_t)row * FEAT;
  for (int i = threadIdx.x; i < FEAT; i += blockDim.x)
    xo[i] = (bf16)((xr[i] - mu) * rstd * g[i] + be[i]);
}

// ---------------------------------------------------------------------------
// Kernel 2: weight fp32 (K x N) -> bf16 transposed (N x K)
// ---------------------------------------------------------------------------
__global__ void transpose_w(const float* __restrict__ W, bf16* __restrict__ WT,
                            int K, int N) {
  int idx = blockIdx.x * blockDim.x + threadIdx.x;
  if (idx >= K * N) return;
  int k = idx / N, n = idx - k * N;
  WT[(size_t)n * K + k] = (bf16)W[idx];
}

// ---------------------------------------------------------------------------
// Kernel 3: generic WMMA GEMM. C[MxN] = A[MxK] * BT[NxK]^T
// One wave computes a 32x64 tile: 2 A-fragments x 4 B-fragments per k-step,
// 8 WMMAs per 12 b128 loads.
// EPI==0: write bf16. EPI==1: write fp32 with +bias[n]+residual[m][n].
// ---------------------------------------------------------------------------
template <int EPI>
__global__ void gemm_bf16(const bf16* __restrict__ A, const bf16* __restrict__ BT,
                          void* __restrict__ Cout, int M, int N, int K,
                          const float* __restrict__ bias,
                          const float* __restrict__ residual) {
  const int wave = (int)((blockIdx.x * blockDim.x + threadIdx.x) >> 5);
  const int lane = threadIdx.x & 31;
  const int ntiles = N >> 6;                 // 64-wide column tiles
  const int mt = wave / ntiles;
  const int nt = wave - mt * ntiles;
  if (mt * 32 >= M) return;
  const int row0 = mt * 32, col0 = nt * 64;

  v8f acc[2][4] = {};
  for (int k0 = 0; k0 < K; k0 += 32) {
    v16bf a0 = load_a_frag(A, K, row0, k0, lane);
    v16bf a1 = load_a_frag(A, K, row0 + 16, k0, lane);
#pragma unroll
    for (int j = 0; j < 4; ++j) {
      v16bf b = load_bt_frag(BT, K, col0 + j * 16, k0, lane);
      acc[0][j] = wmma_bf16(a0, b, acc[0][j]);
      acc[1][j] = wmma_bf16(a1, b, acc[1][j]);
    }
  }

  const int half = lane >> 4, cl = lane & 15;
#pragma unroll
  for (int i = 0; i < 2; ++i) {
#pragma unroll
    for (int j = 0; j < 4; ++j) {
#pragma unroll
      for (int r = 0; r < 8; ++r) {
        const size_t row = (size_t)(row0 + i * 16 + r + 8 * half);
        const size_t col = (size_t)(col0 + j * 16 + cl);
        float v = acc[i][j][r];
        if (EPI == 0) {
          ((bf16*)Cout)[row * N + col] = (bf16)v;
        } else {
          ((float*)Cout)[row * N + col] = v + bias[col] + residual[row * N + col];
        }
      }
    }
  }
}

// ---------------------------------------------------------------------------
// Kernel 4: extract V^T: vT[b][d][s] = kv[b][s][64+d]   (bf16)
// ---------------------------------------------------------------------------
__global__ void extract_vT(const bf16* __restrict__ kvb, bf16* __restrict__ vT) {
  int idx = blockIdx.x * blockDim.x + threadIdx.x;
  if (idx >= BATCH * HDIM * TSEQ) return;
  int s = idx & (TSEQ - 1);
  int d = (idx >> 11) & (HDIM - 1);
  int b = idx >> 17;
  vT[idx] = kvb[((size_t)b * TSEQ + s) * (2 * HDIM) + HDIM + d];
}

// ---------------------------------------------------------------------------
// Kernel 5: flash attention (MQA). One wave per (b, h, 16-row q tile).
//
// Score tiles are computed TRANSPOSED: S'[s][t] = k[s]·q[t].  In C-layout,
// columns = lanes, so each lane's register elements all belong to a single
// query row t = lane&15.  Softmax stats therefore reduce serially in
// registers with a single xor-16 shuffle to merge lane halves; the only
// other cross-lane traffic is broadcasting the per-t correction factor to
// the per-register rows of the O accumulator.  The P'->P transpose for the
// PV matmul is free via the LDS staging (write [t][s], reload in A-layout).
// ---------------------------------------------------------------------------
__global__ void flash_attn(const bf16* __restrict__ qb, const bf16* __restrict__ kvb,
                           const bf16* __restrict__ vT, bf16* __restrict__ attn_out) {
  const int wid = (int)((blockIdx.x * blockDim.x + threadIdx.x) >> 5);
  const int lane = threadIdx.x & 31;
  const int wlocal = threadIdx.x >> 5;
  const int ttile = wid & (TSEQ / 16 - 1);
  const int bh = wid >> 7;                 // / (2048/16)
  const int h = bh & (HEADS - 1);
  const int b = bh >> 4;
  const int t0 = ttile * 16;
  const int half = lane >> 4, cl = lane & 15;

  const bf16* qbase = qb + (size_t)b * TSEQ * FEAT + h * HDIM;  // ld = 1024
  const bf16* kbase = kvb + (size_t)b * TSEQ * (2 * HDIM);      // ld = 128
  const bf16* vtb = vT + (size_t)b * HDIM * TSEQ;               // ld = 2048

  // q as B-fragments (B^T = q rows t, d-contiguous); kept live for whole loop
  const v16bf qf0 = load_bt_frag(qbase, FEAT, t0, 0, lane);   // d 0..31
  const v16bf qf1 = load_bt_frag(qbase, FEAT, t0, 32, lane);  // d 32..63

  // per-wave P staging: [t][s], row stride 40 elems (80 B, 16B-aligned rows)
  constexpr int PLD = 40;
  __shared__ __align__(16) bf16 plds[4][16][PLD];

  float mloc = -1e30f, lloc = 0.f;         // stats for t = lane&15
  v8f O[4] = {};

  for (int s0 = 0; s0 < TSEQ; s0 += 32) {
    if (s0 + 32 < TSEQ) {
      __builtin_prefetch(kbase + (size_t)(s0 + 32) * (2 * HDIM), 0, 1);
      __builtin_prefetch(vtb + (size_t)cl * TSEQ + s0 + 32, 0, 1);
    }
    // S' tiles (s x t): A = k rows (s, d-contiguous), B = q^T
    v8f Sa = {}, Sb = {};
    {
      v16bf ka;
      ka = load_a_frag(kbase, 2 * HDIM, s0, 0, lane);       Sa = wmma_bf16(ka, qf0, Sa);
      ka = load_a_frag(kbase, 2 * HDIM, s0, 32, lane);      Sa = wmma_bf16(ka, qf1, Sa);
      ka = load_a_frag(kbase, 2 * HDIM, s0 + 16, 0, lane);  Sb = wmma_bf16(ka, qf0, Sb);
      ka = load_a_frag(kbase, 2 * HDIM, s0 + 16, 32, lane); Sb = wmma_bf16(ka, qf1, Sb);
    }

    // --- online softmax, stats per lane (one t per lane) ---
    float mx = -1e30f;
#pragma unroll
    for (int r = 0; r < 8; ++r) {
      Sa[r] *= 0.125f;                      // 1/sqrt(64)
      Sb[r] *= 0.125f;
      mx = fmaxf(mx, fmaxf(Sa[r], Sb[r]));
    }
    mx = fmaxf(mx, __shfl_xor(mx, 16, 32)); // merge the two s-halves
    const float mnew = fmaxf(mloc, mx);
    const float corr = __expf(mloc - mnew);
    float rs = 0.f;
#pragma unroll
    for (int r = 0; r < 8; ++r) {
      const float pa = __expf(Sa[r] - mnew);
      const float pb = __expf(Sb[r] - mnew);
      rs += pa + pb;
      // stage P at [t][s]: this lane's elements are (t=cl, s=r+8*half[+16])
      plds[wlocal][cl][r + 8 * half] = (bf16)pa;
      plds[wlocal][cl][16 + r + 8 * half] = (bf16)pb;
    }
    rs += __shfl_xor(rs, 16, 32);
    lloc = lloc * corr + rs;
    mloc = mnew;

    // broadcast corr (held at lane t, duplicated in both halves) to the
    // per-register rows of O: O element r has t = r + 8*half
    float creg[8];
#pragma unroll
    for (int r = 0; r < 8; ++r) creg[r] = __shfl(corr, r + 8 * half, 32);
#pragma unroll
    for (int j = 0; j < 4; ++j)
#pragma unroll
      for (int r = 0; r < 8; ++r) O[j][r] *= creg[r];

    // same-wave DS ops are in-order; ensure stores landed before reload
    asm volatile("s_wait_dscnt 0" ::: "memory");

    const v16bf pA = load_a_frag(&plds[wlocal][0][0], PLD, 0, 0, lane);
#pragma unroll
    for (int j = 0; j < 4; ++j) {
      v16bf bv = load_bt_frag(vtb, TSEQ, j * 16, s0, lane);
      O[j] = wmma_bf16(pA, bv, O[j]);
    }
    asm volatile("s_wait_dscnt 0" ::: "memory");
  }

  // normalize: 1/l held per lane t; broadcast to per-register rows
  const float linv = 1.f / lloc;
  float lreg[8];
#pragma unroll
  for (int r = 0; r < 8; ++r) lreg[r] = __shfl(linv, r + 8 * half, 32);
#pragma unroll
  for (int j = 0; j < 4; ++j)
#pragma unroll
    for (int r = 0; r < 8; ++r) {
      const size_t row = (size_t)b * TSEQ + t0 + r + 8 * half;
      attn_out[row * FEAT + h * HDIM + j * 16 + cl] = (bf16)(O[j][r] * lreg[r]);
    }
}

// ---------------------------------------------------------------------------
// Host-side launch
// ---------------------------------------------------------------------------
extern "C" void kernel_launch(void* const* d_in, const int* in_sizes, int n_in,
                              void* d_out, int out_size, void* d_ws, size_t ws_size,
                              hipStream_t stream) {
  const float* x     = (const float*)d_in[0];
  const float* gamma = (const float*)d_in[1];
  const float* beta  = (const float*)d_in[2];
  const float* Wq    = (const float*)d_in[3];
  const float* Wkv   = (const float*)d_in[4];
  const float* Wo    = (const float*)d_in[5];
  const float* bo    = (const float*)d_in[6];
  float* out = (float*)d_out;

  char* w = (char*)d_ws;
  bf16* xnb   = (bf16*)w; w += (size_t)ROWS * FEAT * 2;        // 8 MB
  bf16* WqT   = (bf16*)w; w += (size_t)FEAT * FEAT * 2;        // 2 MB
  bf16* WkvT  = (bf16*)w; w += (size_t)(2 * HDIM) * FEAT * 2;  // 256 KB
  bf16* WoT   = (bf16*)w; w += (size_t)FEAT * FEAT * 2;        // 2 MB
  bf16* qb    = (bf16*)w; w += (size_t)ROWS * FEAT * 2;        // 8 MB
  bf16* kvb   = (bf16*)w; w += (size_t)ROWS * (2 * HDIM) * 2;  // 1 MB
  bf16* vTt   = (bf16*)w; w += (size_t)BATCH * HDIM * TSEQ * 2;// 512 KB
  bf16* attnb = (bf16*)w; w += (size_t)ROWS * FEAT * 2;        // 8 MB
  (void)ws_size; (void)in_sizes; (void)n_in; (void)out_size;

  // 1. LayerNorm -> bf16
  ln_kernel<<<ROWS, 256, 0, stream>>>(x, gamma, beta, xnb);

  // 2. transpose + convert weights
  transpose_w<<<(FEAT * FEAT + 255) / 256, 256, 0, stream>>>(Wq, WqT, FEAT, FEAT);
  transpose_w<<<(FEAT * 2 * HDIM + 255) / 256, 256, 0, stream>>>(Wkv, WkvT, FEAT, 2 * HDIM);
  transpose_w<<<(FEAT * FEAT + 255) / 256, 256, 0, stream>>>(Wo, WoT, FEAT, FEAT);

  // 3. q = xn @ Wq   (4096x1024x1024): 32x64 tiles -> 2048 waves -> 256 blocks
  gemm_bf16<0><<<256, 256, 0, stream>>>(xnb, WqT, qb, ROWS, FEAT, FEAT, nullptr, nullptr);
  // 4. kv = xn @ Wkv (4096x128x1024): 256 waves -> 32 blocks
  gemm_bf16<0><<<32, 256, 0, stream>>>(xnb, WkvT, kvb, ROWS, 2 * HDIM, FEAT, nullptr, nullptr);

  // 5. V^T for the PV matmul
  extract_vT<<<(BATCH * HDIM * TSEQ + 255) / 256, 256, 0, stream>>>(kvb, vTt);

  // 6. flash attention: 2*16*128 = 4096 waves, 4 waves/block -> 1024 blocks
  flash_attn<<<1024, 128, 0, stream>>>(qb, kvb, vTt, attnb);

  // 7. out = attn @ Wo + bo + x  (fp32 epilogue into d_out)
  gemm_bf16<1><<<256, 256, 0, stream>>>(attnb, WoT, out, ROWS, FEAT, FEAT, bo, x);
}
